// GatRegressionV1_56581899157740
// MI455X (gfx1250) — matrix operand order repaired and checked by reading, not compiled, not run on hardware
//
#include <hip/hip_runtime.h>
#include <string.h>

#define NEG_SLOPE 0.2f

typedef _Float16 v16h __attribute__((ext_vector_type(16)));
typedef float    v8f  __attribute__((ext_vector_type(8)));

// ---------- helpers ----------
__device__ __forceinline__ float lrelu(float a) { return a > 0.f ? a : NEG_SLOPE * a; }
__device__ __forceinline__ float eluf(float a)  { return a > 0.f ? a : (__expf(a) - 1.f); }

// ---------- generic fill ----------
__global__ void __launch_bounds__(256)
fill_u32(unsigned* __restrict__ p, size_t n, unsigned v) {
  size_t i = (size_t)blockIdx.x * blockDim.x + threadIdx.x;
  if (i < n) p[i] = v;
}

// ---------- WMMA GEMM: C[n_rows x 32] = A[n_rows x K] * B[K x 32] (f32 in, f16 mma, f32 acc)
__global__ void __launch_bounds__(256)
gemm_wmma_f16(const float* __restrict__ A, const float* __restrict__ B,
              float* __restrict__ C, int n_rows, int K) {
  int wave = blockIdx.x * (blockDim.x >> 5) + (threadIdx.x >> 5);
  int lane = threadIdx.x & 31;
  int row0 = wave * 16;
  if (row0 >= n_rows) return;          // wave-uniform: EXEC stays all-ones
  int m  = lane & 15;
  int hi = lane >> 4;                  // 0 or 1

  v8f acc0 = {}; v8f acc1 = {};
  for (int k0 = 0; k0 < K; k0 += 32) {
    v16h a, b0, b1;
    const float* arow = A + (size_t)(row0 + m) * K + k0;
    #pragma unroll
    for (int p = 0; p < 8; ++p) {       // A 16-bit 16x32 layout (wave32)
      a[p]     = (_Float16)arow[hi * 8 + p];
      a[p + 8] = (_Float16)arow[16 + hi * 8 + p];
    }
    #pragma unroll
    for (int p = 0; p < 16; ++p) {      // B 32x16: half p <-> K, lane%16 <-> N
      size_t kk = (size_t)(k0 + ((p & 8) << 1) + hi * 8 + (p & 7)) * 32;
      b0[p] = (_Float16)B[kk + m];
      b1[p] = (_Float16)B[kk + 16 + m];
    }
    acc0 = __builtin_amdgcn_wmma_f32_16x16x32_f16(false, a, false, b0,
                                                  (short)0, acc0, false, false);
    acc1 = __builtin_amdgcn_wmma_f32_16x16x32_f16(false, a, false, b1,
                                                  (short)0, acc1, false, false);
  }
  #pragma unroll
  for (int v = 0; v < 8; ++v) {         // D layout: M = v + 8*hi, N = lane%16
    size_t r = (size_t)(row0 + v + hi * 8) * 32;
    C[r + m]      = acc0[v];
    C[r + 16 + m] = acc1[v];
  }
}

// ---------- degree (int) + scatter-sum of edge_attr over original edges ----------
__global__ void __launch_bounds__(256)
deg_attr_kernel(const int* __restrict__ ei, const float* __restrict__ eattr,
                int E0, int* __restrict__ degi, float* __restrict__ ats) {
  int e = blockIdx.x * blockDim.x + threadIdx.x;
  if (e >= E0) return;
  int d = ei[(size_t)E0 + e];
  atomicAdd(&degi[d], 1);
  #pragma unroll
  for (int j = 0; j < 8; ++j)
    atomicAdd(&ats[(size_t)d * 8 + j], eattr[(size_t)e * 8 + j]);
}

// ---------- We1a[k][h] = sum_c We1[k, h*8+c] * a_e1[h,c]   (8x4) ----------
__global__ void we1a_kernel(const float* __restrict__ We1,
                            const float* __restrict__ a_e1,
                            float* __restrict__ wea) {
  int t = threadIdx.x;
  if (t >= 32) return;
  int k = t >> 2, h = t & 3;
  float s = 0.f;
  #pragma unroll
  for (int c = 0; c < 8; ++c) s += We1[k * 32 + h * 8 + c] * a_e1[h * 8 + c];
  wea[k * 4 + h] = s;
}

// ---------- self-loop edge alpha: ael[n,h] = mean_attr[n] . We1a[:,h] ----------
__global__ void __launch_bounds__(256)
aeloop_kernel(const int* __restrict__ degi, const float* __restrict__ ats,
              const float* __restrict__ wea, int N, float* __restrict__ ael) {
  int nid = blockIdx.x * blockDim.x + threadIdx.x;
  if (nid >= N) return;
  float dv = (float)degi[nid]; dv = dv > 1.f ? dv : 1.f;
  float mch[8];
  #pragma unroll
  for (int j = 0; j < 8; ++j) mch[j] = ats[(size_t)nid * 8 + j] / dv;
  #pragma unroll
  for (int h = 0; h < 4; ++h) {
    float s = 0.f;
    #pragma unroll
    for (int j = 0; j < 8; ++j) s += mch[j] * wea[j * 4 + h];
    ael[(size_t)nid * 4 + h] = s;
  }
}

// ---------- CSR build: exclusive scan of (deg+1), 1024 elems/block ----------
__global__ void __launch_bounds__(256)
scan_local(const int* __restrict__ degi, int N, int* __restrict__ off,
           int* __restrict__ bsum) {
  __shared__ int lds[256];
  int base = blockIdx.x * 1024;
  int t = threadIdx.x;
  int v[4]; int s = 0;
  #pragma unroll
  for (int j = 0; j < 4; ++j) {
    int i = base + t * 4 + j;
    v[j] = (i < N) ? (degi[i] + 1) : 0;
    s += v[j];
  }
  lds[t] = s;
  __syncthreads();
  #pragma unroll
  for (int d = 1; d < 256; d <<= 1) {   // Hillis-Steele inclusive scan
    int x = (t >= d) ? lds[t - d] : 0;
    __syncthreads();
    lds[t] += x;
    __syncthreads();
  }
  int excl = (t == 0) ? 0 : lds[t - 1];
  if (t == 255) bsum[blockIdx.x] = lds[255];
  #pragma unroll
  for (int j = 0; j < 4; ++j) {
    int i = base + t * 4 + j;
    if (i < N) off[i] = excl;
    excl += v[j];
  }
}

__global__ void scan_bsum(int* __restrict__ bsum, int nb) {
  if (blockIdx.x == 0 && threadIdx.x == 0) {
    int run = 0;
    for (int i = 0; i < nb; ++i) { int v = bsum[i]; bsum[i] = run; run += v; }
  }
}

__global__ void __launch_bounds__(256)
scan_add(int* __restrict__ off, int N, const int* __restrict__ bsum) {
  int i = blockIdx.x * blockDim.x + threadIdx.x;
  if (i < N) off[i] += bsum[i >> 10];
}

// ---------- CSR fill: slot edges (incl. self-loops) under their dst ----------
__global__ void __launch_bounds__(256)
csr_fill(const int* __restrict__ ei, int E0, int E,
         const int* __restrict__ off, int* __restrict__ cursor,
         int* __restrict__ csr) {
  int e = blockIdx.x * blockDim.x + threadIdx.x;
  if (e >= E) return;
  int d = (e < E0) ? ei[(size_t)E0 + e] : (e - E0);
  int pos = atomicAdd(&cursor[d], 1);
  csr[(size_t)off[d] + pos] = e;
}

// ---------- per-node alpha_src/alpha_dst from xh (H=4, C=8) ----------
__global__ void __launch_bounds__(256)
asd_kernel(const float* __restrict__ xh, const float* __restrict__ a_src,
           const float* __restrict__ a_dst, int N, float* __restrict__ asd) {
  int nid = blockIdx.x * blockDim.x + threadIdx.x;
  if (nid >= N) return;
  float v[32];
  #pragma unroll
  for (int i = 0; i < 32; ++i) v[i] = xh[(size_t)nid * 32 + i];
  #pragma unroll
  for (int h = 0; h < 4; ++h) {
    float ss = 0.f, sd = 0.f;
    #pragma unroll
    for (int c = 0; c < 8; ++c) {
      ss += v[h * 8 + c] * a_src[h * 8 + c];
      sd += v[h * 8 + c] * a_dst[h * 8 + c];
    }
    asd[(size_t)nid * 8 + h]     = ss;
    asd[(size_t)nid * 8 + 4 + h] = sd;
  }
}

// ---------- fused per-node GAT aggregation: online softmax, zero atomics ----------
// layer: 1 = concat+b1+elu (out N*32), 2 = head-mean+b2+elu (out N*8),
//        3 = head-mean+b3 (out N*1)
template <int C, bool USE_EDGE>
__global__ void __launch_bounds__(256)
gat_node_kernel(const int* __restrict__ ei, int E0, int E, int N,
                const int* __restrict__ off, const int* __restrict__ csr,
                const float* __restrict__ asd, const float* __restrict__ ael,
                const float* __restrict__ eattr, const float* __restrict__ wea,
                const float* __restrict__ xh, const float* __restrict__ bias,
                int layer, float* __restrict__ outp) {
  int nid = blockIdx.x * blockDim.x + threadIdx.x;
  if (nid >= N) return;
  int start = off[nid];
  int end   = (nid == N - 1) ? E : off[nid + 1];

  float adst[4], aelv[4];
  #pragma unroll
  for (int h = 0; h < 4; ++h) adst[h] = asd[(size_t)nid * 8 + 4 + h];
  if (USE_EDGE) {
    #pragma unroll
    for (int h = 0; h < 4; ++h) aelv[h] = ael[(size_t)nid * 4 + h];
  }

  float m[4], ssum[4], acc[4 * C];
  #pragma unroll
  for (int h = 0; h < 4; ++h) { m[h] = -3.0e38f; ssum[h] = 0.f; }
  #pragma unroll
  for (int i = 0; i < 4 * C; ++i) acc[i] = 0.f;

  for (int idx = start; idx < end; ++idx) {
    int e = csr[idx];
    int s = (e < E0) ? ei[e] : (e - E0);
    float al[4];
    #pragma unroll
    for (int h = 0; h < 4; ++h) al[h] = asd[(size_t)s * 8 + h] + adst[h];
    if (USE_EDGE) {
      if (e < E0) {
        float ea[8];
        #pragma unroll
        for (int j = 0; j < 8; ++j) ea[j] = eattr[(size_t)e * 8 + j];
        #pragma unroll
        for (int h = 0; h < 4; ++h) {
          float t = 0.f;
          #pragma unroll
          for (int j = 0; j < 8; ++j) t += ea[j] * wea[j * 4 + h];
          al[h] += t;
        }
      } else {
        #pragma unroll
        for (int h = 0; h < 4; ++h) al[h] += aelv[h];
      }
    }
    float xv[4 * C];
    #pragma unroll
    for (int i = 0; i < 4 * C; ++i) xv[i] = xh[(size_t)s * 4 * C + i];
    #pragma unroll
    for (int h = 0; h < 4; ++h) {
      float a = lrelu(al[h]);
      if (a <= m[h]) {                    // online softmax update
        float p = __expf(a - m[h]);
        ssum[h] += p;
        #pragma unroll
        for (int c = 0; c < C; ++c) acc[h * C + c] += p * xv[h * C + c];
      } else {
        float r = __expf(m[h] - a);
        ssum[h] = ssum[h] * r + 1.f;
        #pragma unroll
        for (int c = 0; c < C; ++c)
          acc[h * C + c] = acc[h * C + c] * r + xv[h * C + c];
        m[h] = a;
      }
    }
  }

  if (layer == 1) {
    #pragma unroll
    for (int h = 0; h < 4; ++h) {
      float inv = 1.f / (ssum[h] + 1e-16f);
      #pragma unroll
      for (int c = 0; c < C; ++c)
        outp[(size_t)nid * 4 * C + h * C + c] =
            eluf(acc[h * C + c] * inv + bias[h * C + c]);
    }
  } else if (layer == 2) {
    #pragma unroll
    for (int c = 0; c < C; ++c) {
      float t = 0.f;
      #pragma unroll
      for (int h = 0; h < 4; ++h) t += acc[h * C + c] / (ssum[h] + 1e-16f);
      outp[(size_t)nid * C + c] = eluf(0.25f * t + bias[c]);
    }
  } else {
    float t = 0.f;
    #pragma unroll
    for (int h = 0; h < 4; ++h) t += acc[h * C] / (ssum[h] + 1e-16f);
    outp[nid] = 0.25f * t + bias[0];
  }
}

// ---------- layer-3 prep: xh3 = h2 @ W3 (8->4), per-node alpha terms ----------
__global__ void __launch_bounds__(256)
l3pre_kernel(const float* __restrict__ h2, const float* __restrict__ W3,
             const float* __restrict__ a_src3, const float* __restrict__ a_dst3,
             int N, float* __restrict__ xh3, float* __restrict__ asd) {
  int nid = blockIdx.x * blockDim.x + threadIdx.x;
  if (nid >= N) return;
  float v[8];
  #pragma unroll
  for (int j = 0; j < 8; ++j) v[j] = h2[(size_t)nid * 8 + j];
  #pragma unroll
  for (int h = 0; h < 4; ++h) {
    float s = 0.f;
    #pragma unroll
    for (int j = 0; j < 8; ++j) s += v[j] * W3[j * 4 + h];
    xh3[(size_t)nid * 4 + h]     = s;
    asd[(size_t)nid * 8 + h]     = s * a_src3[h];
    asd[(size_t)nid * 8 + 4 + h] = s * a_dst3[h];
  }
}

// ---------- host ----------
extern "C" void kernel_launch(void* const* d_in, const int* in_sizes, int n_in,
                              void* d_out, int out_size, void* d_ws, size_t ws_size,
                              hipStream_t stream) {
  const float* x      = (const float*)d_in[0];
  const float* eattr  = (const float*)d_in[1];
  const int*   ei     = (const int*)  d_in[2];
  const float* W1     = (const float*)d_in[3];
  const float* a_src1 = (const float*)d_in[4];
  const float* a_dst1 = (const float*)d_in[5];
  const float* We1    = (const float*)d_in[6];
  const float* a_e1   = (const float*)d_in[7];
  const float* b1     = (const float*)d_in[8];
  const float* W2     = (const float*)d_in[9];
  const float* a_src2 = (const float*)d_in[10];
  const float* a_dst2 = (const float*)d_in[11];
  const float* b2     = (const float*)d_in[12];
  const float* W3     = (const float*)d_in[13];
  const float* a_src3 = (const float*)d_in[14];
  const float* a_dst3 = (const float*)d_in[15];
  const float* b3     = (const float*)d_in[16];

  const int N  = in_sizes[0] / 128;
  const int E0 = in_sizes[1] / 8;
  const int E  = E0 + N;               // with self-loops
  const size_t n = (size_t)N;
  const int nb = (N + 1023) / 1024;    // scan blocks

  // workspace layout
  float* F0  = (float*)d_ws;           // N*32 current features
  float* F1  = F0 + 32 * n;            // N*32 xh (layer3: N*4)
  float* ASD = F1 + 32 * n;            // N*8  alpha_src|alpha_dst
  float* AEL = ASD + 8 * n;            // N*4  self-loop edge alpha
  int*   DEGI = (int*)(AEL + 4 * n);   // N    in-degree (int)
  float* ATS = (float*)(DEGI + n);     // N*8  edge_attr scatter sum (zeroed with DEGI)
  float* WEA = ATS + 8 * n;            // 32   We1 . a_e1
  int*   OFF = (int*)(WEA + 32);       // N    CSR offsets
  int*   BSUM = OFF + n;               // nb   scan block sums
  int*   CURS = BSUM + nb;             // N    fill cursors
  int*   CSR  = CURS + n;              // E    edge ids grouped by dst

  const dim3 blk(256);
  const int nodeB  = (N + 255) / 256;
  const int edge0B = (E0 + 255) / 256;
  const int edgeB  = (E + 255) / 256;
  const int gemmB  = (((N + 15) / 16) + 7) / 8;  // 8 waves/block, 16 rows/wave

  // ---- prep: degree, scatter-mean attr, collapsed edge matrix, CSR ----
  fill_u32<<<(int)((9 * n + 255) / 256), blk, 0, stream>>>((unsigned*)DEGI, 9 * n, 0u);
  fill_u32<<<nodeB, blk, 0, stream>>>((unsigned*)CURS, n, 0u);
  deg_attr_kernel<<<edge0B, blk, 0, stream>>>(ei, eattr, E0, DEGI, ATS);
  we1a_kernel<<<1, 32, 0, stream>>>(We1, a_e1, WEA);
  aeloop_kernel<<<nodeB, blk, 0, stream>>>(DEGI, ATS, WEA, N, AEL);
  scan_local<<<nb, blk, 0, stream>>>(DEGI, N, OFF, BSUM);
  scan_bsum<<<1, 1, 0, stream>>>(BSUM, nb);
  scan_add<<<nodeB, blk, 0, stream>>>(OFF, N, BSUM);
  csr_fill<<<edgeB, blk, 0, stream>>>(ei, E0, E, OFF, CURS, CSR);

  // ---- layer 1: x@W1 (WMMA), fused gather/online-softmax/ELU ----
  gemm_wmma_f16<<<gemmB, blk, 0, stream>>>(x, W1, F1, N, 128);
  asd_kernel<<<nodeB, blk, 0, stream>>>(F1, a_src1, a_dst1, N, ASD);
  gat_node_kernel<8, true><<<nodeB, blk, 0, stream>>>(
      ei, E0, E, N, OFF, CSR, ASD, AEL, eattr, WEA, F1, b1, 1, F0);

  // ---- layer 2: h1@W2 (WMMA), fused aggregation + head-mean + ELU ----
  gemm_wmma_f16<<<gemmB, blk, 0, stream>>>(F0, W2, F1, N, 32);
  asd_kernel<<<nodeB, blk, 0, stream>>>(F1, a_src2, a_dst2, N, ASD);
  gat_node_kernel<8, false><<<nodeB, blk, 0, stream>>>(
      ei, E0, E, N, OFF, CSR, ASD, AEL, eattr, WEA, F1, b2, 2, F0);

  // ---- layer 3: tiny 8->4 transform, fused aggregation + head-mean ----
  l3pre_kernel<<<nodeB, blk, 0, stream>>>(F0, W3, a_src3, a_dst3, N, F1, ASD);
  gat_node_kernel<1, false><<<nodeB, blk, 0, stream>>>(
      ei, E0, E, N, OFF, CSR, ASD, AEL, eattr, WEA, F1, b3, 3, (float*)d_out);
}
